// PatchedTransformer_66340064854634
// MI455X (gfx1250) — matrix operand order repaired and checked by reading, not compiled, not run on hardware
//
#include <hip/hip_runtime.h>
#include <hip/hip_bf16.h>

// ---------------- problem constants ----------------
constexpr int CB  = 2;      // batch
constexpr int CL  = 1024;   // input tokens
constexpr int CD  = 1024;   // model dim
constexpr int CH  = 16;     // heads
constexpr int CDH = 64;     // head dim
constexpr int CFD = 512;    // input feature dim
constexpr int CNL = 4;      // layers
constexpr int CNV = 5000;   // vocab
constexpr int CDF = 4096;   // ffn dim
constexpr int CSV = 1026;   // valid seq len (1 graph + 1 sos + 1024)
constexpr int CSP = 1056;   // padded seq len (multiple of 32)
constexpr int CNVP = 5120;  // vocab padded to 128 (block-tile N)

typedef float  v8f   __attribute__((ext_vector_type(8)));
typedef __bf16 v16bf __attribute__((ext_vector_type(16)));

union FragU { v16bf v; uint4 q[2]; };

// fp32 -> bf16 (RNE) as raw ushort
__device__ __forceinline__ unsigned short f2bfu(float f) {
    unsigned u = __float_as_uint(f);
    return (unsigned short)((u + 0x7FFFu + ((u >> 16) & 1u)) >> 16);
}

// A-fragment (16x32 bf16, M across lanes, ISA 7.12.2 layout):
// lane half h: chunks at k0 + h*8 and k0 + 16 + h*8
__device__ __forceinline__ v16bf ld_fragA(const unsigned short* p, int ld, int row, int k0, int lane) {
    int r = lane & 15, hf = lane >> 4;
    const unsigned short* a = p + (size_t)(row + r) * (size_t)ld + (size_t)(k0 + hf * 8);
    FragU f;
    f.q[0] = *(const uint4*)a;
    f.q[1] = *(const uint4*)(a + 16);
    return f.v;
}

// B-fragment (32x16 bf16, N across lanes, same K layout as A-fragment):
// source is W^T stored N x K row-major -> chunks at k0 + half*8 and k0 + 16 + half*8
__device__ __forceinline__ v16bf ld_fragB(const unsigned short* p, int ld, int row, int k0, int lane) {
    int r = lane & 15, hf = lane >> 4;
    const unsigned short* a = p + (size_t)(row + r) * (size_t)ld + (size_t)(k0 + hf * 8);
    FragU f;
    f.q[0] = *(const uint4*)a;
    f.q[1] = *(const uint4*)(a + 16);
    return f.v;
}

__device__ __forceinline__ v8f wmma_bf16(v16bf a, v16bf b, v8f c) {
    return __builtin_amdgcn_wmma_f32_16x16x32_bf16(false, a, false, b, (short)0, c, false, false);
}

// ---------------- elementwise fp32 -> bf16 ----------------
__global__ void k_f2b(const float* __restrict__ in, unsigned short* __restrict__ out, int n) {
    int i = blockIdx.x * 256 + threadIdx.x;
    if (i < n) out[i] = f2bfu(in[i]);
}

// ---------------- LDS-tiled transpose + convert: W (K x N) fp32 -> Wt (Npad x K) bf16 ----------------
__global__ __launch_bounds__(256) void k_wT(const float* __restrict__ W, unsigned short* __restrict__ Wt,
                                            int K, int N, int Npad) {
    __shared__ float tile[32][33];
    int n0 = blockIdx.x * 32, k0 = blockIdx.y * 32;
    int tx = threadIdx.x, ty = threadIdx.y;
#pragma unroll
    for (int j = 0; j < 4; j++) {
        int k = k0 + ty + 8 * j, n = n0 + tx;
        tile[ty + 8 * j][tx] = (n < N) ? W[(size_t)k * N + n] : 0.f;
    }
    __syncthreads();
#pragma unroll
    for (int j = 0; j < 4; j++) {
        int n = n0 + ty + 8 * j, k = k0 + tx;
        Wt[(size_t)n * K + k] = f2bfu(tile[tx][ty + 8 * j]);
    }
    (void)Npad;
}

// ---------------- embeddings / token assembly into h (fp32, CB*CSP x CD) ----------------
__global__ __launch_bounds__(256) void k_embed(float* __restrict__ h, const float* __restrict__ graph,
                                               const float* __restrict__ sos, const float* __restrict__ node_emb,
                                               const float* __restrict__ type_emb, const float* __restrict__ pos_emb,
                                               const int* __restrict__ sub) {
    int row = blockIdx.x;
    int b = row / CSP, s = row % CSP;
    float* hr = h + (size_t)row * CD;
    int t = threadIdx.x;
#pragma unroll
    for (int i = 0; i < 4; i++) {
        int d = t + 256 * i;
        float val;
        if (s == 0)        val = graph[(size_t)b * CD + d] + type_emb[CD + d] + pos_emb[d];
        else if (s == 1)   val = sos[d] + node_emb[d] + type_emb[d] + pos_emb[CD + d];
        else if (s < CSV)  val = hr[d] + node_emb[(size_t)sub[b * CL + (s - 2)] * CD + d]
                                 + type_emb[d] + pos_emb[(size_t)s * CD + d];
        else               val = 0.f;
        hr[d] = val;
    }
}

// ---------------- RMSNorm: h fp32 -> x bf16, zero pad rows ----------------
__global__ __launch_bounds__(256) void k_rmsnorm(const float* __restrict__ h, const float* __restrict__ w,
                                                 unsigned short* __restrict__ x) {
    int row = blockIdx.x;
    int s = row % CSP;
    int t = threadIdx.x;
    const float* hr = h + (size_t)row * CD;
    unsigned short* xr = x + (size_t)row * CD;
    if (s >= CSV) {
#pragma unroll
        for (int i = 0; i < 4; i++) xr[t + 256 * i] = 0;
        return;
    }
    __shared__ float red[8];
    float vals[4], ss = 0.f;
#pragma unroll
    for (int i = 0; i < 4; i++) { vals[i] = hr[t + 256 * i]; ss += vals[i] * vals[i]; }
#pragma unroll
    for (int off = 1; off < 32; off <<= 1) ss += __shfl_xor(ss, off, 32);
    if ((t & 31) == 0) red[t >> 5] = ss;
    __syncthreads();
    float tot = 0.f;
#pragma unroll
    for (int i = 0; i < 8; i++) tot += red[i];
    float rs = rsqrtf(tot * (1.0f / CD) + 1e-6f);
#pragma unroll
    for (int i = 0; i < 4; i++) xr[t + 256 * i] = f2bfu(vals[i] * rs * w[t + 256 * i]);
}

// ---------------- V transpose: v (b,s,h*64+dh) -> vT (b,h,dh, s) ----------------
__global__ void k_vT(const unsigned short* __restrict__ v, unsigned short* __restrict__ vT) {
    int idx = blockIdx.x * 256 + threadIdx.x;
    int c = idx & (CD - 1);
    int rest = idx >> 10;
    int s = rest % CSP, b = rest / CSP;
    int hh = c >> 6, dh = c & 63;
    vT[((size_t)((b * CH + hh) * CDH + dh)) * CSP + s] = v[idx];
}

// ---------------- generic WMMA GEMM: C(MxN) = A(MxK) * Bt(NxK)^T ----------------
// wave tile 32x64 (2 A-frags x 4 B-frags -> 8 WMMAs / K-step), block tile 64x128 (4 waves)
// MODE 0: Cf = acc (+bias)   1: Cb = bf16(acc)   2: Cf += acc
// MODE 3: Cb = bf16(silu(aux)*acc)   4: final vocab proj -> outF with row remap
template <int MODE>
__global__ __launch_bounds__(128) void k_gemm(const unsigned short* __restrict__ A,
                                              const unsigned short* __restrict__ Bt,
                                              float* __restrict__ Cf, unsigned short* __restrict__ Cb,
                                              const float* __restrict__ aux, const float* __restrict__ bias,
                                              float* __restrict__ outF,
                                              int K, int lda, int ldb, int ldc) {
    int lane = threadIdx.x & 31;
    int wave = threadIdx.x >> 5;
    int m0 = blockIdx.y * 64 + (wave >> 1) * 32;
    int n0 = blockIdx.x * 128 + (wave & 1) * 64;
    v8f acc[2][4] = {};
    for (int k0 = 0; k0 < K; k0 += 32) {
        v16bf a0 = ld_fragA(A, lda, m0, k0, lane);
        v16bf a1 = ld_fragA(A, lda, m0 + 16, k0, lane);
        v16bf b0 = ld_fragB(Bt, ldb, n0, k0, lane);
        v16bf b1 = ld_fragB(Bt, ldb, n0 + 16, k0, lane);
        v16bf b2 = ld_fragB(Bt, ldb, n0 + 32, k0, lane);
        v16bf b3 = ld_fragB(Bt, ldb, n0 + 48, k0, lane);
        acc[0][0] = wmma_bf16(a0, b0, acc[0][0]);
        acc[0][1] = wmma_bf16(a0, b1, acc[0][1]);
        acc[0][2] = wmma_bf16(a0, b2, acc[0][2]);
        acc[0][3] = wmma_bf16(a0, b3, acc[0][3]);
        acc[1][0] = wmma_bf16(a1, b0, acc[1][0]);
        acc[1][1] = wmma_bf16(a1, b1, acc[1][1]);
        acc[1][2] = wmma_bf16(a1, b2, acc[1][2]);
        acc[1][3] = wmma_bf16(a1, b3, acc[1][3]);
    }
    int r = lane & 15, hf = lane >> 4;
#pragma unroll
    for (int i = 0; i < 2; i++)
#pragma unroll
        for (int j = 0; j < 4; j++) {
            int n = n0 + j * 16 + r;
#pragma unroll
            for (int v = 0; v < 8; v++) {
                int m = m0 + i * 16 + hf * 8 + v;
                float val = acc[i][j][v];
                size_t idx = (size_t)m * ldc + n;
                if constexpr (MODE == 0) {
                    Cf[idx] = val + (bias ? bias[n] : 0.f);
                } else if constexpr (MODE == 1) {
                    Cb[idx] = f2bfu(val);
                } else if constexpr (MODE == 2) {
                    Cf[idx] += val;
                } else if constexpr (MODE == 3) {
                    float g = aux[idx];
                    float sg = g / (1.f + __expf(-g));
                    Cb[idx] = f2bfu(sg * val);
                } else {  // MODE 4: final vocab projection
                    int bb = m / CSP, s = m % CSP;
                    if (s >= 1 && s < CSV && n < CNV)
                        outF[((size_t)bb * (CSV - 1) + (s - 1)) * CNV + n] = val + bias[n];
                }
            }
        }
}

// ---------------- SPD bias + query-axis mask ----------------
__device__ __forceinline__ float bias_mask(int qq, int kk, const int* __restrict__ spd_b,
                                           const float* __restrict__ s_spd, int tlim) {
    if (kk >= CSV) return -1.0e30f;  // padded key: contributes exp()=0
    int idx;
    if (qq >= 2 && qq < CSV && kk >= 2) idx = spd_b[(qq - 2) * CL + (kk - 2)] & 31;
    else idx = (qq == kk && qq < 2) ? 0 : 31;
    float m = (qq < tlim) ? 0.f : -1.0e9f;  // reference masks the QUERY axis
    return s_spd[idx] + m;
}

// ---------------- flash attention: one wave per (b, head, 16-query tile) ----------------
__global__ __launch_bounds__(32) void k_attn(const unsigned short* __restrict__ q,
                                             const unsigned short* __restrict__ kmat,
                                             const unsigned short* __restrict__ vT,
                                             unsigned short* __restrict__ att,
                                             const float* __restrict__ spd_emb,
                                             const int* __restrict__ spd_idx,
                                             const int* __restrict__ tml) {
    int qt = blockIdx.x, hh = blockIdx.y, b = blockIdx.z;
    int lane = threadIdx.x;
    __shared__ float s_spd[32];
    __shared__ __align__(16) unsigned short s_p[16 * 32];
    s_spd[lane] = spd_emb[lane * CH + hh];
    int tlim = tml[b] + 2;
    const int* spd_b = spd_idx + (size_t)b * CL * CL;
    __syncthreads();

    const unsigned short* qb = q + (size_t)b * CSP * CD;
    const unsigned short* kb = kmat + (size_t)b * CSP * CD;
    const unsigned short* vb = vT + (size_t)(b * CH + hh) * CDH * CSP;

    int q0 = qt * 16;
    v16bf qa0 = ld_fragA(qb, CD, q0, hh * 64, lane);
    v16bf qa1 = ld_fragA(qb, CD, q0, hh * 64 + 32, lane);

    v8f O0 = {}, O1 = {}, O2 = {}, O3 = {};
    v8f mrow, lrow;
#pragma unroll
    for (int v = 0; v < 8; v++) { mrow[v] = -3.0e38f; lrow[v] = 0.f; }
    int r = lane & 15, hf = lane >> 4;

    for (int kt = 0; kt < CSP; kt += 32) {
        // scores for 32 keys: Q is A, K rows (seq x 64) are already N x K for B-fragments
        v16bf kb0a = ld_fragB(kb, CD, kt, hh * 64, lane);
        v16bf kb0b = ld_fragB(kb, CD, kt, hh * 64 + 32, lane);
        v16bf kb1a = ld_fragB(kb, CD, kt + 16, hh * 64, lane);
        v16bf kb1b = ld_fragB(kb, CD, kt + 16, hh * 64 + 32, lane);
        v8f S0 = {}, S1 = {};
        S0 = wmma_bf16(qa0, kb0a, S0);
        S0 = wmma_bf16(qa1, kb0b, S0);
        S1 = wmma_bf16(qa0, kb1a, S1);
        S1 = wmma_bf16(qa1, kb1b, S1);

        int key0 = kt + r, key1 = kt + 16 + r;
#pragma unroll
        for (int v = 0; v < 8; v++) {
            int qq = q0 + hf * 8 + v;
            S0[v] = S0[v] * 0.125f + bias_mask(qq, key0, spd_b, s_spd, tlim);
            S1[v] = S1[v] * 0.125f + bias_mask(qq, key1, spd_b, s_spd, tlim);
        }
        // online softmax: rows live inside one 16-lane half -> xor-shuffle reduce
        v8f bm;
#pragma unroll
        for (int v = 0; v < 8; v++) bm[v] = fmaxf(S0[v], S1[v]);
#pragma unroll
        for (int off = 1; off < 16; off <<= 1)
#pragma unroll
            for (int v = 0; v < 8; v++) bm[v] = fmaxf(bm[v], __shfl_xor(bm[v], off, 32));
        v8f sc;
#pragma unroll
        for (int v = 0; v < 8; v++) {
            float mn = fmaxf(mrow[v], bm[v]);
            sc[v] = __expf(mrow[v] - mn);
            mrow[v] = mn;
            S0[v] = __expf(S0[v] - mn);
            S1[v] = __expf(S1[v] - mn);
        }
        v8f rs;
#pragma unroll
        for (int v = 0; v < 8; v++) rs[v] = S0[v] + S1[v];
#pragma unroll
        for (int off = 1; off < 16; off <<= 1)
#pragma unroll
            for (int v = 0; v < 8; v++) rs[v] += __shfl_xor(rs[v], off, 32);
#pragma unroll
        for (int v = 0; v < 8; v++) {
            lrow[v] = lrow[v] * sc[v] + rs[v];
            O0[v] *= sc[v]; O1[v] *= sc[v]; O2[v] *= sc[v]; O3[v] *= sc[v];
        }
        // re-layout P (C-layout regs) -> LDS -> A-fragment
        __syncthreads();
#pragma unroll
        for (int v = 0; v < 8; v++) {
            int row = v + 8 * hf;
            s_p[row * 32 + r] = f2bfu(S0[v]);
            s_p[row * 32 + 16 + r] = f2bfu(S1[v]);
        }
        __syncthreads();
        v16bf pa = ld_fragA(s_p, 32, 0, 0, lane);
        v16bf vb0 = ld_fragB(vb, CSP, 0, kt, lane);
        v16bf vb1 = ld_fragB(vb, CSP, 16, kt, lane);
        v16bf vb2 = ld_fragB(vb, CSP, 32, kt, lane);
        v16bf vb3 = ld_fragB(vb, CSP, 48, kt, lane);
        O0 = wmma_bf16(pa, vb0, O0);
        O1 = wmma_bf16(pa, vb1, O1);
        O2 = wmma_bf16(pa, vb2, O2);
        O3 = wmma_bf16(pa, vb3, O3);
    }
#pragma unroll
    for (int v = 0; v < 8; v++) {
        int qrow = q0 + v + 8 * hf;
        float inv = lrow[v] > 0.f ? 1.f / lrow[v] : 0.f;
        size_t base = (size_t)(b * CSP + qrow) * CD + hh * 64;
        bool ok = qrow < CSV;
        att[base + r]      = ok ? f2bfu(O0[v] * inv) : (unsigned short)0;
        att[base + 16 + r] = ok ? f2bfu(O1[v] * inv) : (unsigned short)0;
        att[base + 32 + r] = ok ? f2bfu(O2[v] * inv) : (unsigned short)0;
        att[base + 48 + r] = ok ? f2bfu(O3[v] * inv) : (unsigned short)0;
    }
}

// ---------------- host orchestration ----------------
extern "C" void kernel_launch(void* const* d_in, const int* in_sizes, int n_in,
                              void* d_out, int out_size, void* d_ws, size_t ws_size,
                              hipStream_t stream) {
    const float* graph    = (const float*)d_in[0];
    const float* inputf   = (const float*)d_in[1];
    const float* projW    = (const float*)d_in[2];
    const float* projB    = (const float*)d_in[3];
    const float* sos      = (const float*)d_in[4];
    const float* node_emb = (const float*)d_in[5];
    const float* type_emb = (const float*)d_in[6];
    const float* pos_emb  = (const float*)d_in[7];
    const float* spd_emb  = (const float*)d_in[8];
    const float* attn_nw  = (const float*)d_in[9];
    const float* wq       = (const float*)d_in[10];
    const float* wk       = (const float*)d_in[11];
    const float* wv       = (const float*)d_in[12];
    const float* wo       = (const float*)d_in[13];
    const float* ffn_nw   = (const float*)d_in[14];
    const float* w1       = (const float*)d_in[15];
    const float* w2       = (const float*)d_in[16];
    const float* w3       = (const float*)d_in[17];
    const float* final_nw = (const float*)d_in[18];
    const float* outW     = (const float*)d_in[19];
    const float* outB     = (const float*)d_in[20];
    const int*   sub      = (const int*)d_in[21];
    const int*   tml      = (const int*)d_in[22];
    const int*   spd_idx  = (const int*)d_in[23];
    float* out = (float*)d_out;

    char* base = (char*)d_ws;
    size_t off = 0;
    auto alloc = [&](size_t bytes) -> void* {
        void* p = base + off;
        off = (off + bytes + 255) & ~(size_t)255;
        return p;
    };
    const int MR = CB * CSP;  // 2112 padded activation rows
    unsigned short* x_in   = (unsigned short*)alloc((size_t)CB * CL * CFD * 2);
    unsigned short* projWt = (unsigned short*)alloc((size_t)CD * CFD * 2);
    float*          h      = (float*)alloc((size_t)MR * CD * 4);
    unsigned short* xb     = (unsigned short*)alloc((size_t)MR * CD * 2);
    unsigned short* qb     = (unsigned short*)alloc((size_t)MR * CD * 2);
    unsigned short* kbuf   = (unsigned short*)alloc((size_t)MR * CD * 2);
    unsigned short* vbuf   = (unsigned short*)alloc((size_t)MR * CD * 2);
    unsigned short* vTb    = (unsigned short*)alloc((size_t)MR * CD * 2);
    unsigned short* attb   = (unsigned short*)alloc((size_t)MR * CD * 2);
    unsigned short* wqT    = (unsigned short*)alloc((size_t)CD * CD * 2);
    unsigned short* wkT    = (unsigned short*)alloc((size_t)CD * CD * 2);
    unsigned short* wvT    = (unsigned short*)alloc((size_t)CD * CD * 2);
    unsigned short* woT    = (unsigned short*)alloc((size_t)CD * CD * 2);
    unsigned short* w1T    = (unsigned short*)alloc((size_t)CDF * CD * 2);
    unsigned short* w3T    = (unsigned short*)alloc((size_t)CDF * CD * 2);
    unsigned short* w2T    = (unsigned short*)alloc((size_t)CD * CDF * 2);
    unsigned short* outWt  = (unsigned short*)alloc((size_t)CNVP * CD * 2);
    float*          gbuf   = (float*)alloc((size_t)MR * CDF * 4);
    unsigned short* ybuf   = (unsigned short*)alloc((size_t)MR * CDF * 2);
    (void)in_sizes; (void)n_in; (void)out_size; (void)ws_size;

    // input features -> bf16; project into h rows s=2..1025 (per batch, affine row map)
    int ntot = CB * CL * CFD;
    k_f2b<<<(ntot + 255) / 256, 256, 0, stream>>>(inputf, x_in, ntot);
    k_wT<<<dim3(CD / 32, CFD / 32), dim3(32, 8), 0, stream>>>(projW, projWt, CFD, CD, CD);
    for (int b = 0; b < CB; b++)
        k_gemm<0><<<dim3(CD / 128, CL / 64), 128, 0, stream>>>(
            x_in + (size_t)b * CL * CFD, projWt, h + ((size_t)b * CSP + 2) * CD,
            nullptr, nullptr, projB, nullptr, CFD, CFD, CFD, CD);
    k_embed<<<MR, 256, 0, stream>>>(h, graph, sos, node_emb, type_emb, pos_emb, sub);

    for (int i = 0; i < CNL; i++) {
        // attention
        k_rmsnorm<<<MR, 256, 0, stream>>>(h, attn_nw + (size_t)i * CD, xb);
        k_wT<<<dim3(CD / 32, CD / 32), dim3(32, 8), 0, stream>>>(wq + (size_t)i * CD * CD, wqT, CD, CD, CD);
        k_wT<<<dim3(CD / 32, CD / 32), dim3(32, 8), 0, stream>>>(wk + (size_t)i * CD * CD, wkT, CD, CD, CD);
        k_wT<<<dim3(CD / 32, CD / 32), dim3(32, 8), 0, stream>>>(wv + (size_t)i * CD * CD, wvT, CD, CD, CD);
        k_wT<<<dim3(CD / 32, CD / 32), dim3(32, 8), 0, stream>>>(wo + (size_t)i * CD * CD, woT, CD, CD, CD);
        k_gemm<1><<<dim3(CD / 128, MR / 64), 128, 0, stream>>>(xb, wqT, nullptr, qb, nullptr, nullptr, nullptr, CD, CD, CD, CD);
        k_gemm<1><<<dim3(CD / 128, MR / 64), 128, 0, stream>>>(xb, wkT, nullptr, kbuf, nullptr, nullptr, nullptr, CD, CD, CD, CD);
        k_gemm<1><<<dim3(CD / 128, MR / 64), 128, 0, stream>>>(xb, wvT, nullptr, vbuf, nullptr, nullptr, nullptr, CD, CD, CD, CD);
        k_vT<<<(MR * CD) / 256, 256, 0, stream>>>(vbuf, vTb);
        k_attn<<<dim3(CSP / 16, CH, CB), 32, 0, stream>>>(qb, kbuf, vTb, attb, spd_emb, spd_idx, tml);
        k_gemm<2><<<dim3(CD / 128, MR / 64), 128, 0, stream>>>(attb, woT, h, nullptr, nullptr, nullptr, nullptr, CD, CD, CD, CD);
        // ffn
        k_rmsnorm<<<MR, 256, 0, stream>>>(h, ffn_nw + (size_t)i * CD, xb);
        k_wT<<<dim3(CDF / 32, CD / 32), dim3(32, 8), 0, stream>>>(w1 + (size_t)i * CD * CDF, w1T, CD, CDF, CDF);
        k_wT<<<dim3(CDF / 32, CD / 32), dim3(32, 8), 0, stream>>>(w3 + (size_t)i * CD * CDF, w3T, CD, CDF, CDF);
        k_wT<<<dim3(CD / 32, CDF / 32), dim3(32, 8), 0, stream>>>(w2 + (size_t)i * CDF * CD, w2T, CDF, CD, CD);
        k_gemm<0><<<dim3(CDF / 128, MR / 64), 128, 0, stream>>>(xb, w1T, gbuf, nullptr, nullptr, nullptr, nullptr, CD, CD, CD, CDF);
        k_gemm<3><<<dim3(CDF / 128, MR / 64), 128, 0, stream>>>(xb, w3T, nullptr, ybuf, gbuf, nullptr, nullptr, CD, CD, CD, CDF);
        k_gemm<2><<<dim3(CD / 128, MR / 64), 128, 0, stream>>>(ybuf, w2T, h, nullptr, nullptr, nullptr, nullptr, CDF, CDF, CDF, CD);
    }

    // final norm + vocab projection straight into d_out
    k_rmsnorm<<<MR, 256, 0, stream>>>(h, final_nw, xb);
    k_wT<<<dim3(CNVP / 32, CD / 32), dim3(32, 8), 0, stream>>>(outW, outWt, CD, CNV, CNVP);
    k_gemm<4><<<dim3(CNVP / 128, MR / 64), 128, 0, stream>>>(xb, outWt, nullptr, nullptr, nullptr, outB, out, CD, CD, CD, CNVP);
}
